// LSTM_30288109371765
// MI455X (gfx1250) — compile-verified, matrix operand order checked by reading, not run on hardware
//
#include <hip/hip_runtime.h>
#include <hip/hip_bf16.h>

typedef __attribute__((ext_vector_type(16))) __bf16 v16bf;
typedef __attribute__((ext_vector_type(8)))  __bf16 v8bf;
typedef __attribute__((ext_vector_type(8)))  float  v8f;

#define Bsz   512
#define Tsz   1024
#define INsz  11
#define Hsz   60
#define Gsz   240          // 4*H gates
#define NTILES 15          // 240/16
#define KPAD0 96           // pad(11+60) -> 3 K-chunks of 32
#define KPAD1 192          // pad(120+60) -> 6 K-chunks of 32
#define MT    16           // batch-rows per workgroup
#define NWAVES 5           // 5 waves x 3 tiles = 15 tiles, fully static
#define TPW   3            // tiles per wave
#define BLOCK 160          // 5 waves of 32

// Per-lane 32B fragment load (two 16B chunks; contiguous-K A/B order used here (A and B match):
// lane%16 = row (A) / col (B), lane>>4 = K-half of the 32-wide K chunk)
__device__ __forceinline__ v16bf load_frag(const __bf16* p) {
  v8bf lo = *(const v8bf*)p;
  v8bf hi = *(const v8bf*)(p + 8);
  return __builtin_shufflevector(lo, hi, 0,1,2,3,4,5,6,7,8,9,10,11,12,13,14,15);
}

__device__ __forceinline__ float sigf(float x) { return 1.0f / (1.0f + __expf(-x)); }

// ---------------------------------------------------------------------------
// Pack combined [Wih | Whh] (K-padded, bf16, row-major over gates) + fused bias
// ---------------------------------------------------------------------------
__global__ void prep_kernel(
    const float* wih0f, const float* whh0f, const float* bih0f, const float* bhh0f,
    const float* wih0b, const float* whh0b, const float* bih0b, const float* bhh0b,
    const float* wih1f, const float* whh1f, const float* bih1f, const float* bhh1f,
    const float* wih1b, const float* whh1b, const float* bih1b, const float* bhh1b,
    __bf16* W0f, __bf16* W0b, __bf16* W1f, __bf16* W1b, float* bias)
{
  int idx = blockIdx.x * blockDim.x + threadIdx.x;
  if (idx < Gsz * KPAD0) {
    int n = idx / KPAD0, k = idx % KPAD0;
    float vf = 0.f, vb = 0.f;
    if (k < INsz)            { vf = wih0f[n*INsz + k];        vb = wih0b[n*INsz + k]; }
    else if (k < INsz + Hsz) { vf = whh0f[n*Hsz + (k-INsz)];  vb = whh0b[n*Hsz + (k-INsz)]; }
    W0f[idx] = (__bf16)vf;  W0b[idx] = (__bf16)vb;
  }
  if (idx < Gsz * KPAD1) {
    int n = idx / KPAD1, k = idx % KPAD1;
    float vf = 0.f, vb = 0.f;
    if (k < 2*Hsz)           { vf = wih1f[n*(2*Hsz) + k];      vb = wih1b[n*(2*Hsz) + k]; }
    else if (k < 3*Hsz)      { vf = whh1f[n*Hsz + (k-2*Hsz)];  vb = whh1b[n*Hsz + (k-2*Hsz)]; }
    W1f[idx] = (__bf16)vf;  W1b[idx] = (__bf16)vb;
  }
  if (idx < Gsz) {
    bias[0*Gsz + idx] = bih0f[idx] + bhh0f[idx];
    bias[1*Gsz + idx] = bih0b[idx] + bhh0b[idx];
    bias[2*Gsz + idx] = bih1f[idx] + bhh1f[idx];
    bias[3*Gsz + idx] = bih1b[idx] + bhh1b[idx];
  }
}

// ---------------------------------------------------------------------------
// Layer 0: full bidirectional scan. 64 WGs = 32 batch tiles x 2 directions.
// Weights + h/c state resident in LDS; 2 barriers per timestep.
// ---------------------------------------------------------------------------
__global__ void lstm_l0_kernel(const float* __restrict__ data,
                               const __bf16* __restrict__ Wf,
                               const __bf16* __restrict__ Wb,
                               const float* __restrict__ bias,
                               __bf16* __restrict__ y0)
{
  extern __shared__ char smem[];
  __bf16* Wl = (__bf16*)smem;              // 240*96 bf16  (46 KB)
  __bf16* Al = Wl + Gsz * KPAD0;           // 16*96 bf16   ([x|h|pad] per row)
  float*  bl = (float*)(Al + MT * KPAD0);  // 240 f32
  float*  gl = bl + Gsz;                   // 16*240 f32 gate staging
  float*  cl = gl + MT * Gsz;              // 16*60 f32 cell state

  const int tileB = blockIdx.x & 31;
  const int dir   = blockIdx.x >> 5;
  const int b0    = tileB * MT;
  const __bf16* Wg = dir ? Wb : Wf;
  const float*  bg = bias + dir * Gsz;

  for (int i = threadIdx.x; i < Gsz * KPAD0 / 8; i += BLOCK)
    ((v8bf*)Wl)[i] = ((const v8bf*)Wg)[i];
  for (int i = threadIdx.x; i < Gsz; i += BLOCK)        bl[i] = bg[i];
  for (int i = threadIdx.x; i < MT * KPAD0; i += BLOCK)  Al[i] = (__bf16)0.0f;
  for (int i = threadIdx.x; i < MT * Hsz; i += BLOCK)    cl[i] = 0.0f;
  __syncthreads();

  const int lane = threadIdx.x & 31;
  const int wave = threadIdx.x >> 5;
  const int nm   = lane & 15;   // row of A / col of B within a tile
  const int kh   = lane >> 4;   // K-half selector
  const int t0   = wave * TPW, t1 = t0 + 1, t2 = t0 + 2;

  const __bf16* Ap  = Al + nm * KPAD0 + kh * 16;            // + 32*kc
  const __bf16* Wp0 = Wl + (t0 * 16 + nm) * KPAD0 + kh * 16;
  const __bf16* Wp1 = Wl + (t1 * 16 + nm) * KPAD0 + kh * 16;
  const __bf16* Wp2 = Wl + (t2 * 16 + nm) * KPAD0 + kh * 16;
  float* gp0 = gl + (8 * kh) * Gsz + t0 * 16 + nm;          // + r*Gsz
  float* gp1 = gl + (8 * kh) * Gsz + t1 * 16 + nm;
  float* gp2 = gl + (8 * kh) * Gsz + t2 * 16 + nm;

  // prologue: stage x at the first timestep
  {
    const int tf = dir ? (Tsz - 1) : 0;
    for (int u = threadIdx.x; u < MT * INsz; u += BLOCK) {
      int m = u / INsz, k = u % INsz;
      Al[m * KPAD0 + k] = (__bf16)data[((size_t)(b0 + m) * Tsz + tf) * INsz + k];
    }
  }
  __syncthreads();

  for (int s = 0; s < Tsz; ++s) {
    const int t = dir ? (Tsz - 1 - s) : s;

    v8f acc0 = {}, acc1 = {}, acc2 = {};
    #pragma unroll
    for (int kc = 0; kc < KPAD0 / 32; ++kc) {
      v16bf a  = load_frag(Ap  + kc * 32);
      v16bf w0 = load_frag(Wp0 + kc * 32);
      v16bf w1 = load_frag(Wp1 + kc * 32);
      v16bf w2 = load_frag(Wp2 + kc * 32);
      acc0 = __builtin_amdgcn_wmma_f32_16x16x32_bf16(false, a, false, w0, (short)0, acc0, false, false);
      acc1 = __builtin_amdgcn_wmma_f32_16x16x32_bf16(false, a, false, w1, (short)0, acc1, false, false);
      acc2 = __builtin_amdgcn_wmma_f32_16x16x32_bf16(false, a, false, w2, (short)0, acc2, false, false);
    }
    {
      float bv0 = bl[t0 * 16 + nm], bv1 = bl[t1 * 16 + nm], bv2 = bl[t2 * 16 + nm];
      #pragma unroll
      for (int r = 0; r < 8; ++r) {  // C layout: M = r + 8*kh, N = tile*16 + nm
        gp0[r * Gsz] = acc0[r] + bv0;
        gp1[r * Gsz] = acc1[r] + bv1;
        gp2[r * Gsz] = acc2[r] + bv2;
      }
    }
    __syncthreads();

    // gates + state update; also stage x for the next timestep (disjoint Al region)
    for (int u = threadIdx.x; u < MT * Hsz; u += BLOCK) {
      int m = u / Hsz, j = u % Hsz;
      const float* gr = gl + m * Gsz;
      float iv = sigf(gr[j]);
      float fv = sigf(gr[Hsz + j]);
      float gv = tanhf(gr[2 * Hsz + j]);
      float ov = sigf(gr[3 * Hsz + j]);
      float c  = fv * cl[u] + iv * gv;
      float h  = ov * tanhf(c);
      cl[u] = c;
      Al[m * KPAD0 + INsz + j] = (__bf16)h;   // next step's A matrix
      y0[((size_t)(b0 + m) * Tsz + t) * (2 * Hsz) + dir * Hsz + j] = (__bf16)h;
    }
    if (s + 1 < Tsz) {
      const int tn = dir ? (Tsz - 2 - s) : (s + 1);
      for (int u = threadIdx.x; u < MT * INsz; u += BLOCK) {
        int m = u / INsz, k = u % INsz;
        Al[m * KPAD0 + k] = (__bf16)data[((size_t)(b0 + m) * Tsz + tn) * INsz + k];
      }
    }
    __syncthreads();
  }
}

// ---------------------------------------------------------------------------
// Layer 1: forward = full scan keeping only final h; backward = exactly ONE
// step (y1[:, -1] for the reverse direction is the first reverse-scan output).
// ---------------------------------------------------------------------------
__global__ void lstm_l1_kernel(const __bf16* __restrict__ y0,
                               const __bf16* __restrict__ Wf,
                               const __bf16* __restrict__ Wb,
                               const float* __restrict__ bias,
                               float* __restrict__ last)
{
  extern __shared__ char smem[];
  __bf16* Wl = (__bf16*)smem;              // 240*192 bf16 (92 KB in 320KB LDS)
  __bf16* Al = Wl + Gsz * KPAD1;           // 16*192 bf16
  float*  bl = (float*)(Al + MT * KPAD1);
  float*  gl = bl + Gsz;
  float*  cl = gl + MT * Gsz;

  const int tileB = blockIdx.x & 31;
  const int dir   = blockIdx.x >> 5;
  const int b0    = tileB * MT;
  const __bf16* Wg = dir ? Wb : Wf;
  const float*  bg = bias + (2 + dir) * Gsz;

  for (int i = threadIdx.x; i < Gsz * KPAD1 / 8; i += BLOCK)
    ((v8bf*)Wl)[i] = ((const v8bf*)Wg)[i];
  for (int i = threadIdx.x; i < Gsz; i += BLOCK)         bl[i] = bg[i];
  for (int i = threadIdx.x; i < MT * KPAD1; i += BLOCK)   Al[i] = (__bf16)0.0f;
  for (int i = threadIdx.x; i < MT * Hsz; i += BLOCK)     cl[i] = 0.0f;
  __syncthreads();

  const int lane = threadIdx.x & 31;
  const int wave = threadIdx.x >> 5;
  const int nm   = lane & 15;
  const int kh   = lane >> 4;
  const int t0   = wave * TPW, t1 = t0 + 1, t2 = t0 + 2;

  const __bf16* Ap  = Al + nm * KPAD1 + kh * 16;
  const __bf16* Wp0 = Wl + (t0 * 16 + nm) * KPAD1 + kh * 16;
  const __bf16* Wp1 = Wl + (t1 * 16 + nm) * KPAD1 + kh * 16;
  const __bf16* Wp2 = Wl + (t2 * 16 + nm) * KPAD1 + kh * 16;
  float* gp0 = gl + (8 * kh) * Gsz + t0 * 16 + nm;
  float* gp1 = gl + (8 * kh) * Gsz + t1 * 16 + nm;
  float* gp2 = gl + (8 * kh) * Gsz + t2 * 16 + nm;

  const int steps = dir ? 1 : Tsz;

  // prologue: stage x at the first timestep (16 rows x 120 bf16, 16B vectors)
  {
    const int tf = dir ? (Tsz - 1) : 0;
    for (int u = threadIdx.x; u < MT * 15; u += BLOCK) {
      int m = u / 15, ch = u % 15;
      const v8bf* src = (const v8bf*)(y0 + ((size_t)(b0 + m) * Tsz + tf) * (2 * Hsz));
      ((v8bf*)(Al + m * KPAD1))[ch] = src[ch];
    }
  }
  __syncthreads();

  for (int s = 0; s < steps; ++s) {
    v8f acc0 = {}, acc1 = {}, acc2 = {};
    #pragma unroll
    for (int kc = 0; kc < KPAD1 / 32; ++kc) {
      v16bf a  = load_frag(Ap  + kc * 32);
      v16bf w0 = load_frag(Wp0 + kc * 32);
      v16bf w1 = load_frag(Wp1 + kc * 32);
      v16bf w2 = load_frag(Wp2 + kc * 32);
      acc0 = __builtin_amdgcn_wmma_f32_16x16x32_bf16(false, a, false, w0, (short)0, acc0, false, false);
      acc1 = __builtin_amdgcn_wmma_f32_16x16x32_bf16(false, a, false, w1, (short)0, acc1, false, false);
      acc2 = __builtin_amdgcn_wmma_f32_16x16x32_bf16(false, a, false, w2, (short)0, acc2, false, false);
    }
    {
      float bv0 = bl[t0 * 16 + nm], bv1 = bl[t1 * 16 + nm], bv2 = bl[t2 * 16 + nm];
      #pragma unroll
      for (int r = 0; r < 8; ++r) {
        gp0[r * Gsz] = acc0[r] + bv0;
        gp1[r * Gsz] = acc1[r] + bv1;
        gp2[r * Gsz] = acc2[r] + bv2;
      }
    }
    __syncthreads();

    const bool emit = (s == steps - 1);
    for (int u = threadIdx.x; u < MT * Hsz; u += BLOCK) {
      int m = u / Hsz, j = u % Hsz;
      const float* gr = gl + m * Gsz;
      float iv = sigf(gr[j]);
      float fv = sigf(gr[Hsz + j]);
      float gv = tanhf(gr[2 * Hsz + j]);
      float ov = sigf(gr[3 * Hsz + j]);
      float c  = fv * cl[u] + iv * gv;
      float h  = ov * tanhf(c);
      cl[u] = c;
      Al[m * KPAD1 + 2 * Hsz + j] = (__bf16)h;
      if (emit) last[(size_t)(b0 + m) * (2 * Hsz) + dir * Hsz + j] = h;  // f32
    }
    if (s + 1 < steps) {
      const int tn = s + 1;  // only the forward direction iterates
      for (int u = threadIdx.x; u < MT * 15; u += BLOCK) {
        int m = u / 15, ch = u % 15;
        const v8bf* src = (const v8bf*)(y0 + ((size_t)(b0 + m) * Tsz + tn) * (2 * Hsz));
        ((v8bf*)(Al + m * KPAD1))[ch] = src[ch];
        if (tn + 1 < Tsz) __builtin_prefetch((const void*)(src + 15), 0, 0);
      }
    }
    __syncthreads();
  }
}

// ---------------------------------------------------------------------------
// Head: out[b] = (last[b] @ w_lin^T + b_lin) @ w_out^T + b_out   (tiny)
// ---------------------------------------------------------------------------
__global__ void head_kernel(const float* __restrict__ last,
                            const float* __restrict__ wlin, const float* __restrict__ blin,
                            const float* __restrict__ wout, const float* __restrict__ bout,
                            float* __restrict__ out)
{
  int b = blockIdx.x * blockDim.x + threadIdx.x;
  if (b >= Bsz) return;
  const float* L = last + (size_t)b * (2 * Hsz);
  float acc = bout[0];
  for (int q = 0; q < 64; ++q) {
    const float* w = wlin + q * (2 * Hsz);
    float s = blin[q];
    #pragma unroll 4
    for (int k = 0; k < 2 * Hsz; ++k) s += L[k] * w[k];
    acc += wout[q] * s;
  }
  out[b] = acc;
}

extern "C" void kernel_launch(void* const* d_in, const int* in_sizes, int n_in,
                              void* d_out, int out_size, void* d_ws, size_t ws_size,
                              hipStream_t stream)
{
  (void)in_sizes; (void)n_in; (void)out_size; (void)ws_size;
  const float* data = (const float*)d_in[0];

  char* ws = (char*)d_ws;
  size_t off = 0;
  auto take = [&](size_t bytes) -> char* {
    char* p = ws + off;
    off = (off + bytes + 255) & ~(size_t)255;
    return p;
  };
  __bf16* W0f  = (__bf16*)take((size_t)Gsz * KPAD0 * 2);
  __bf16* W0b  = (__bf16*)take((size_t)Gsz * KPAD0 * 2);
  __bf16* W1f  = (__bf16*)take((size_t)Gsz * KPAD1 * 2);
  __bf16* W1b  = (__bf16*)take((size_t)Gsz * KPAD1 * 2);
  float*  bias = (float*)take((size_t)4 * Gsz * 4);
  float*  last = (float*)take((size_t)Bsz * 2 * Hsz * 4);
  __bf16* y0   = (__bf16*)take((size_t)Bsz * Tsz * 2 * Hsz * 2);   // ~126 MB

  prep_kernel<<<(Gsz * KPAD1 + 255) / 256, 256, 0, stream>>>(
      (const float*)d_in[1],  (const float*)d_in[2],  (const float*)d_in[3],  (const float*)d_in[4],
      (const float*)d_in[5],  (const float*)d_in[6],  (const float*)d_in[7],  (const float*)d_in[8],
      (const float*)d_in[9],  (const float*)d_in[10], (const float*)d_in[11], (const float*)d_in[12],
      (const float*)d_in[13], (const float*)d_in[14], (const float*)d_in[15], (const float*)d_in[16],
      W0f, W0b, W1f, W1b, bias);

  size_t sh0 = (size_t)(Gsz * KPAD0 + MT * KPAD0) * 2 + (size_t)(Gsz + MT * Gsz + MT * Hsz) * 4;
  lstm_l0_kernel<<<64, BLOCK, sh0, stream>>>(data, W0f, W0b, bias, y0);

  size_t sh1 = (size_t)(Gsz * KPAD1 + MT * KPAD1) * 2 + (size_t)(Gsz + MT * Gsz + MT * Hsz) * 4;
  lstm_l1_kernel<<<64, BLOCK, sh1, stream>>>(y0, W1f, W1b, bias, last);

  head_kernel<<<(Bsz + 255) / 256, 256, 0, stream>>>(
      last, (const float*)d_in[17], (const float*)d_in[18],
      (const float*)d_in[19], (const float*)d_in[20], (float*)d_out);
}
